// GeodesicStateModule_33036888440842
// MI455X (gfx1250) — matrix operand order repaired and checked by reading, not compiled, not run on hardware
//
#include <hip/hip_runtime.h>
#include <hip/hip_bf16.h>

// ---------------------------------------------------------------------------
// GeodesicStateModule for MI455X (gfx1250, wave32, WMMA).
// GEMMs via v_wmma_f32_16x16x32_bf16 with explicit ping-pong software
// pipelining (loads for chunk k+32 issued before WMMAs of chunk k);
// scan via wave32 ds_swizzle reductions.
// ---------------------------------------------------------------------------

#define B_SZ   4
#define T_LEN  8192
#define D_DIM  512
#define E_DIM  1024
#define N_DIM  64
#define NP     80                 // N_DIM + 1 (delta row) padded to 5*16
#define M_TOT  (B_SZ * T_LEN)     // 32768 tokens
#define ALPHA  0.5f
#define EPS_N  1e-6f

typedef __bf16 bf16_t;
typedef __attribute__((ext_vector_type(16))) __bf16 v16bf;
typedef __attribute__((ext_vector_type(8)))  __bf16 v8bf;
typedef __attribute__((ext_vector_type(8)))  float  v8f;

// ---------------------------------------------------------------------------
// fp32 -> bf16 conversion
// ---------------------------------------------------------------------------
__global__ void cvt_kernel(const float* __restrict__ in, bf16_t* __restrict__ out, int n) {
  int i = blockIdx.x * blockDim.x + threadIdx.x;
  if (i < n) out[i] = (bf16_t)in[i];
}

// Build padded [W_B ; W_delta ; zeros] (NP x E) bf16 weight + its fp32 bias.
__global__ void build_wbd_kernel(const float* __restrict__ W_B, const float* __restrict__ W_delta,
                                 const float* __restrict__ b_B, const float* __restrict__ b_delta,
                                 bf16_t* __restrict__ Wbd, float* __restrict__ bbd) {
  int i = blockIdx.x * blockDim.x + threadIdx.x;
  if (i < NP * E_DIM) {
    int r = i / E_DIM, c = i - r * E_DIM;
    float v = (r < N_DIM) ? W_B[r * E_DIM + c] : ((r == N_DIM) ? W_delta[c] : 0.f);
    Wbd[i] = (bf16_t)v;
  }
  if (i < NP) bbd[i] = (i < N_DIM) ? b_B[i] : ((i == N_DIM) ? b_delta[0] : 0.f);
}

// ---------------------------------------------------------------------------
// Generic WMMA GEMM:  C[M x Nc] = act(A[M x Kd](bf16) * W[Nc x Kd](bf16)^T + bias)
// NTILES: 16-wide N tiles per wave (compile-time; grid.y * 16*NTILES == Nc).
// ACT: SiLU epilogue. MUL: multiply by fp32 tensor. WF/WB: fp32/bf16 outputs.
// Block: 256 threads = 8 waves; each wave -> 16(M) x 16*NTILES(N) C strip.
// Requires Kd % 64 == 0 (all call sites: 512, 1024, 64).
// ---------------------------------------------------------------------------
template<int NTILES, int ACT, int MUL, int WF, int WB>
__global__ __launch_bounds__(256)
void gemm_bf16_wmma(const bf16_t* __restrict__ A, const bf16_t* __restrict__ W,
                    const float* __restrict__ bias, const float* __restrict__ mulin,
                    float* __restrict__ outF, bf16_t* __restrict__ outB,
                    int M, int Nc, int Kd) {
  const int lane = threadIdx.x & 31;
  const int wave = threadIdx.x >> 5;
  const int m0 = blockIdx.x * 128 + wave * 16;
  const int n0 = blockIdx.y * (16 * NTILES);

  // 32-bit element offsets off uniform bases -> saddr + voffset addressing.
  // A fragment (16x32 bf16, MxK): lane L: row = m0 + (L&15);
  // elems 0..7 = K[h*8 .. h*8+7], elems 8..15 = K[16+h*8 ..], h = L>>4.
  const int aoff = (m0 + (lane & 15)) * Kd + (lane >> 4) * 8;
  // B fragment (32x16 bf16, KxN) from row-major W[Nc x Kd] (K contiguous):
  // lane L: col = n + (L&15); elems 0..7 = K[g*8 ..], elems 8..15 = K[16+g*8 ..], g = L>>4.
  int woff[NTILES];
#pragma unroll
  for (int t = 0; t < NTILES; ++t)
    woff[t] = (n0 + t * 16 + (lane & 15)) * Kd + (lane >> 4) * 8;

  union Frag { v16bf v; v8bf h[2]; };
  Frag a0, a1, b0[NTILES], b1[NTILES];
  v8f acc[NTILES] = {};

  auto loadA = [&](Frag& a, int k) {
    a.h[0] = *(const v8bf*)(A + (aoff + k));
    a.h[1] = *(const v8bf*)(A + (aoff + k + 16));
  };
  auto loadB = [&](Frag* b, int k) {
#pragma unroll
    for (int t = 0; t < NTILES; ++t) {
      b[t].h[0] = *(const v8bf*)(W + (woff[t] + k));
      b[t].h[1] = *(const v8bf*)(W + (woff[t] + k + 16));
    }
  };
  auto mac = [&](const Frag& a, const Frag* b) {
#pragma unroll
    for (int t = 0; t < NTILES; ++t)
      acc[t] = __builtin_amdgcn_wmma_f32_16x16x32_bf16(
          false, a.v, false, b[t].v, (short)0, acc[t], false, false);
  };

  // Ping-pong software pipeline over K chunks of 32 (Kd/32 is even).
  loadA(a0, 0);
  loadB(b0, 0);
  for (int k = 0; k < Kd; k += 64) {
    loadA(a1, k + 32);                         // prefetch chunk k+32
    loadB(b1, k + 32);
    __builtin_prefetch(A + aoff + k + 512, 0, 1);  // global_prefetch_b8
    mac(a0, b0);                               // compute chunk k
    const int kn = (k + 64 < Kd) ? (k + 64) : 0;   // clamp (redundant last load)
    loadA(a0, kn);                             // prefetch chunk k+64
    loadB(b0, kn);
    mac(a1, b1);                               // compute chunk k+32
  }

  // C layout: VGPR r, lanes 0..15 -> row r; lanes 16..31 -> row 8+r; col = lane&15.
  const int cl  = lane & 15;
  const int rhi = (lane >> 4) * 8;
#pragma unroll
  for (int t = 0; t < NTILES; ++t) {
    int n = n0 + t * 16 + cl;
    float bv = bias ? bias[n] : 0.f;
#pragma unroll
    for (int r = 0; r < 8; ++r) {
      int ro = m0 + rhi + r;
      float c = acc[t][r] + bv;
      if (ACT) c = c / (1.f + expf(-c));    // SiLU
      size_t idx = (size_t)ro * Nc + n;
      if (MUL) c *= mulin[idx];
      if (WF) outF[idx] = c;
      if (WB) outB[idx] = (bf16_t)c;
    }
  }
}

// ---------------------------------------------------------------------------
// Depthwise conv (K=3, same padding along T) + bias + SiLU -> bf16 x_tilde
// ---------------------------------------------------------------------------
__global__ void conv_silu_kernel(const float* __restrict__ xe, const float* __restrict__ Wc,
                                 const float* __restrict__ bc, bf16_t* __restrict__ xt) {
  size_t i = (size_t)blockIdx.x * blockDim.x + threadIdx.x;
  if (i >= (size_t)M_TOT * E_DIM) return;
  int e = (int)(i & (E_DIM - 1));
  size_t row = i >> 10;                     // / E_DIM
  int t = (int)(row & (T_LEN - 1));         // row = b*T + t
  float xm = (t > 0)         ? xe[i - E_DIM] : 0.f;
  float xc = xe[i];
  float xp = (t < T_LEN - 1) ? xe[i + E_DIM] : 0.f;
  float y = xm * Wc[e * 3 + 0] + xc * Wc[e * 3 + 1] + xp * Wc[e * 3 + 2] + bc[e];
  xt[i] = (bf16_t)(y / (1.f + expf(-y)));
}

// ---------------------------------------------------------------------------
// wave32 XOR-butterfly sum via ds_swizzle_b32 (group-of-32 xor masks)
// ---------------------------------------------------------------------------
__device__ __forceinline__ float wred32(float x) {
  x += __int_as_float(__builtin_amdgcn_ds_swizzle(__float_as_int(x), 0x401f)); // SWAPX16
  x += __int_as_float(__builtin_amdgcn_ds_swizzle(__float_as_int(x), 0x201f)); // SWAPX8
  x += __int_as_float(__builtin_amdgcn_ds_swizzle(__float_as_int(x), 0x101f)); // SWAPX4
  x += __int_as_float(__builtin_amdgcn_ds_swizzle(__float_as_int(x), 0x081f)); // SWAPX2
  x += __int_as_float(__builtin_amdgcn_ds_swizzle(__float_as_int(x), 0x041f)); // SWAPX1
  return x;
}

// ---------------------------------------------------------------------------
// Sequential geodesic scan: 1 wave per batch, 2 state elements per lane.
// UD: (B,T,NP) fp32 with cols 0..63 = U (incl. b_B), col 64 = delta pre-act.
// ---------------------------------------------------------------------------
__global__ void scan_kernel(const float* __restrict__ UD, const float* __restrict__ h0,
                            bf16_t* __restrict__ H) {
  const int b = blockIdx.x;
  const int lane = threadIdx.x;             // 0..31
  float ha = h0[lane], hb = h0[lane + 32];
  float nn = wred32(ha * ha + hb * hb);
  float inv = 1.f / (sqrtf(nn) + EPS_N);
  ha *= inv; hb *= inv;
  float va = 0.f, vb = 0.f;
  const float* ud = UD + (size_t)b * T_LEN * NP;
  bf16_t* hp = H + (size_t)b * T_LEN * N_DIM;
  for (int t = 0; t < T_LEN; ++t) {
    float ua = ud[lane], ub = ud[lane + 32];
    float dp = ud[N_DIM];
    ud += NP;
    float d = (dp > 20.f) ? dp : log1pf(expf(dp));    // softplus
    va = ALPHA * va + ua;
    vb = ALPHA * vb + ub;
    float dot = wred32(va * ha + vb * hb);            // <v, h>
    float ta = va - dot * ha, tb = vb - dot * hb;     // tangent projection
    float na = ha + d * ta,  nb = hb + d * tb;
    float s = wred32(na * na + nb * nb);
    float r = 1.f / (sqrtf(s) + EPS_N);               // retraction to sphere
    ha = na * r; hb = nb * r;
    hp[lane] = (bf16_t)ha; hp[lane + 32] = (bf16_t)hb;
    hp += N_DIM;
  }
}

// ---------------------------------------------------------------------------
// Host orchestration (graph-capture safe: only kernel launches on stream)
// ---------------------------------------------------------------------------
extern "C" void kernel_launch(void* const* d_in, const int* in_sizes, int n_in,
                              void* d_out, int out_size, void* d_ws, size_t ws_size,
                              hipStream_t stream) {
  (void)in_sizes; (void)n_in; (void)out_size; (void)ws_size;
  const float* x       = (const float*)d_in[0];
  const float* W_in    = (const float*)d_in[1];
  const float* b_in    = (const float*)d_in[2];
  const float* W_conv  = (const float*)d_in[3];
  const float* b_conv  = (const float*)d_in[4];
  const float* W_B     = (const float*)d_in[5];
  const float* b_B     = (const float*)d_in[6];
  const float* W_delta = (const float*)d_in[7];
  const float* b_delta = (const float*)d_in[8];
  const float* W_C     = (const float*)d_in[9];
  const float* b_C     = (const float*)d_in[10];
  const float* h0      = (const float*)d_in[11];
  const float* W_gate  = (const float*)d_in[12];
  const float* b_gate  = (const float*)d_in[13];
  const float* W_out   = (const float*)d_in[14];
  const float* b_out   = (const float*)d_in[15];
  float* y = (float*)d_out;

  char* ws = (char*)d_ws;
  size_t off = 0;
  auto carve = [&](size_t bytes) -> char* {
    char* p = ws + off;
    off += (bytes + 255) & ~(size_t)255;
    return p;
  };
  bf16_t* Xb     = (bf16_t*)carve((size_t)M_TOT * D_DIM * 2);   // x in bf16
  bf16_t* Winb   = (bf16_t*)carve((size_t)E_DIM * D_DIM * 2);
  bf16_t* Wbdb   = (bf16_t*)carve((size_t)NP * E_DIM * 2);      // [W_B;W_delta;0]
  bf16_t* Wgateb = (bf16_t*)carve((size_t)E_DIM * E_DIM * 2);
  bf16_t* WCb    = (bf16_t*)carve((size_t)E_DIM * N_DIM * 2);
  bf16_t* Woutb  = (bf16_t*)carve((size_t)D_DIM * E_DIM * 2);
  float*  bbd    = (float*) carve((size_t)NP * 4);
  float*  bufF   = (float*) carve((size_t)M_TOT * E_DIM * 4);   // x_exp, then gate
  bf16_t* Xtb    = (bf16_t*)carve((size_t)M_TOT * E_DIM * 2);   // x_tilde, then z
  float*  UD     = (float*) carve((size_t)M_TOT * NP * 4);      // [U | delta_pre]
  bf16_t* Hb     = (bf16_t*)carve((size_t)M_TOT * N_DIM * 2);   // scan states

  auto cvt = [&](const float* in, bf16_t* out, int n) {
    cvt_kernel<<<(n + 255) / 256, 256, 0, stream>>>(in, out, n);
  };
  cvt(x,      Xb,     M_TOT * D_DIM);
  cvt(W_in,   Winb,   E_DIM * D_DIM);
  cvt(W_gate, Wgateb, E_DIM * E_DIM);
  cvt(W_C,    WCb,    E_DIM * N_DIM);
  cvt(W_out,  Woutb,  D_DIM * E_DIM);
  build_wbd_kernel<<<(NP * E_DIM + 255) / 256, 256, 0, stream>>>(W_B, W_delta, b_B, b_delta, Wbdb, bbd);

  const dim3 blk(256);
  // 1) x_exp = x * W_in^T + b_in   (fp32 out)
  gemm_bf16_wmma<4,0,0,1,0><<<dim3(M_TOT / 128, E_DIM / 64), blk, 0, stream>>>(
      Xb, Winb, b_in, nullptr, bufF, nullptr, M_TOT, E_DIM, D_DIM);
  // 2) depthwise conv + SiLU -> x_tilde (bf16)
  conv_silu_kernel<<<(unsigned)(((size_t)M_TOT * E_DIM + 255) / 256), blk, 0, stream>>>(
      bufF, W_conv, b_conv, Xtb);
  // 3) [U | delta_pre] = x_tilde * [W_B;W_delta]^T + [b_B;b_delta]  (5x16 = 80 cols)
  gemm_bf16_wmma<5,0,0,1,0><<<dim3(M_TOT / 128, 1), blk, 0, stream>>>(
      Xtb, Wbdb, bbd, nullptr, UD, nullptr, M_TOT, NP, E_DIM);
  // 4) geodesic scan (sequential in T; B=4 independent waves)
  scan_kernel<<<dim3(B_SZ), dim3(32), 0, stream>>>(UD, h0, Hb);
  // 5) gate = SiLU(x_tilde * W_gate^T + b_gate)   (fp32 out, reuse bufF)
  gemm_bf16_wmma<4,1,0,1,0><<<dim3(M_TOT / 128, E_DIM / 64), blk, 0, stream>>>(
      Xtb, Wgateb, b_gate, nullptr, bufF, nullptr, M_TOT, E_DIM, E_DIM);
  // 6) z = gate * (H * W_C^T + b_C)   (bf16 out, reuse Xtb)
  gemm_bf16_wmma<4,0,1,0,1><<<dim3(M_TOT / 128, E_DIM / 64), blk, 0, stream>>>(
      Hb, WCb, b_C, bufF, nullptr, Xtb, M_TOT, E_DIM, N_DIM);
  // 7) y = z * W_out^T + b_out
  gemm_bf16_wmma<4,0,0,1,0><<<dim3(M_TOT / 128, D_DIM / 64), blk, 0, stream>>>(
      Xtb, Woutb, b_out, nullptr, y, nullptr, M_TOT, D_DIM, E_DIM);
}
